// LSTMEncoder_54528904790309
// MI455X (gfx1250) — matrix operand order repaired and checked by reading, not compiled, not run on hardware
//
#include <hip/hip_runtime.h>

typedef __attribute__((ext_vector_type(16))) _Float16 v16h;
typedef __attribute__((ext_vector_type(8)))  _Float16 v8h;
typedef __attribute__((ext_vector_type(4)))  _Float16 v4h;
typedef __attribute__((ext_vector_type(8)))  float    v8f;
typedef __attribute__((ext_vector_type(4)))  float    v4f;
typedef __attribute__((ext_vector_type(2)))  float    v2f;

#define TT 25
#define II 4
#define HH 64
#define NG 256          // 4*HH gate width
#define ROWS_PER_WAVE 16
#define WAVES_PER_BLOCK 8
#define ROWS_PER_BLOCK (ROWS_PER_WAVE * WAVES_PER_BLOCK)

// Native CDNA5 transcendental tanh (V_TANH_F32). One trans op instead of
// exp+rcp (two trans ops) per activation.
__device__ __forceinline__ float vtanh(float x) {
#if __has_builtin(__builtin_amdgcn_tanhf)
    return __builtin_amdgcn_tanhf(x);
#else
    float y;
    // v_nop covers the TRANS32 co-execution hazard (1 independent op before use)
    asm("v_tanh_f32 %0, %1\n\tv_nop" : "=v"(y) : "v"(x));
    return y;
#endif
}
__device__ __forceinline__ float sigf(float x) {
    return __builtin_fmaf(0.5f, vtanh(0.5f * x), 0.5f);
}

// B-operand (32x16, f16) from LDS matrix stored row-major [N][64] (N=gate idx, col=K).
// B order used here (contiguous): lanes 0-15 hold K=0..15 (elements 0..15), lanes 16-31 hold K=16..31.
__device__ __forceinline__ v16h lds_b16(const _Float16* base, int n0, int li, int half, int k0) {
    return *(const v16h*)(base + (n0 + li) * HH + k0 + 16 * half);
}

// A-operand (16x32, f16) from LDS matrix stored row-major [M][64].
// A layout (matches B): lane L(<16): M=L, elems 0-15 = K k0+0..15;
// lane L+16: M=L, elems 0-15 = K k0+16..31.
__device__ __forceinline__ v16h lds_a16(const _Float16* base, int li, int half, int k0) {
    return *(const v16h*)(base + li * HH + k0 + 16 * half);
}

__global__ __launch_bounds__(256) void lstm2_wmma_kernel(
    const float* __restrict__ x,      // [B, T, 4]
    const float* __restrict__ Wih0,   // [256, 4]
    const float* __restrict__ Whh0,   // [256, 64]
    const float* __restrict__ b0g,    // [256]
    const float* __restrict__ Wih1,   // [256, 64]
    const float* __restrict__ Whh1,   // [256, 64]
    const float* __restrict__ b1g,    // [256]
    float* __restrict__ out)          // [B, 5, 64]
{
    __shared__ _Float16 sWhh0[NG * HH];   // 32 KB, f16 [n][k]
    __shared__ _Float16 sWih1[NG * HH];   // 32 KB
    __shared__ _Float16 sWhh1[NG * HH];   // 32 KB
    __shared__ float    sWih0[NG * II];   // 4 KB, f32 [n][k]
    __shared__ float    sB0[NG];
    __shared__ float    sB1[NG];
    __shared__ _Float16 sHbuf[WAVES_PER_BLOCK][2][ROWS_PER_WAVE * HH]; // 32 KB

    const int tid = threadIdx.x;

    // ---- Stage weights into LDS (f32 -> f16 for the K=64 matrices) ----
    {
        const int N = NG * HH;
        for (int i = tid * 4; i < N; i += 256 * 4) {
            v4f w0 = *(const v4f*)(Whh0 + i);
            v4f w1 = *(const v4f*)(Wih1 + i);
            v4f w2 = *(const v4f*)(Whh1 + i);
            v4h h0, h1v, h2v;
#pragma unroll
            for (int e = 0; e < 4; ++e) {
                h0[e]  = (_Float16)w0[e];
                h1v[e] = (_Float16)w1[e];
                h2v[e] = (_Float16)w2[e];
            }
            *(v4h*)(sWhh0 + i) = h0;
            *(v4h*)(sWih1 + i) = h1v;
            *(v4h*)(sWhh1 + i) = h2v;
        }
        for (int i = tid; i < NG * II; i += 256) sWih0[i] = Wih0[i];
        for (int i = tid; i < NG; i += 256) { sB0[i] = b0g[i]; sB1[i] = b1g[i]; }
    }
    __syncthreads();

    const int wave = tid >> 5;
    const int lane = tid & 31;
    const int li   = lane & 15;   // row/col within 16
    const int half = lane >> 4;   // 0 or 1

    const long brow = ((long)blockIdx.x * WAVES_PER_BLOCK + wave) * ROWS_PER_WAVE;
    _Float16* h1buf = &sHbuf[wave][0][0];
    _Float16* h2buf = &sHbuf[wave][1][0];

    // Recurrent state, resident in VGPRs (C-layout tiles: VGPR r -> row m = r + 8*half, col = 16j+li)
    v8f c1[4], c2[4];
#pragma unroll
    for (int j = 0; j < 4; ++j) {
        c1[j] = (v8f){0.f, 0.f, 0.f, 0.f, 0.f, 0.f, 0.f, 0.f};
        c2[j] = (v8f){0.f, 0.f, 0.f, 0.f, 0.f, 0.f, 0.f, 0.f};
    }
    v16h a1lo = (v16h)(_Float16)0.0f, a1hi = (v16h)(_Float16)0.0f;
    v16h a2lo = (v16h)(_Float16)0.0f, a2hi = (v16h)(_Float16)0.0f;

    // x A-operand base: A 16x4 f32 layout -> lane(li,half): M=li, K = 2*half + {0,1}
    const float* xrow = x + (size_t)(brow + li) * (TT * II) + 2 * half;

#pragma unroll 1
    for (int t = 0; t < TT; ++t) {
        v2f ax = *(const v2f*)(xrow + t * II);

        // ================= Layer 1 =================
#pragma unroll
        for (int j = 0; j < 4; ++j) {
            v8f acc[4];
#pragma unroll
            for (int g = 0; g < 4; ++g) {       // gate order: i, f, g, o
                const int n0 = g * HH + j * 16;
                const float bv = sB0[n0 + li];
                v8f a = {bv, bv, bv, bv, bv, bv, bv, bv};
                // x @ W_ih0^T : one fp32 WMMA, K = 4
                v2f bx = *(const v2f*)(sWih0 + (n0 + li) * II + 2 * half);
                a = __builtin_amdgcn_wmma_f32_16x16x4_f32(false, ax, false, bx,
                                                          (short)0, a, false, false);
                // h1 @ W_hh0^T : two f16 WMMAs, K = 64
                a = __builtin_amdgcn_wmma_f32_16x16x32_f16(false, a1lo, false,
                        lds_b16(sWhh0, n0, li, half, 0), (short)0, a, false, false);
                a = __builtin_amdgcn_wmma_f32_16x16x32_f16(false, a1hi, false,
                        lds_b16(sWhh0, n0, li, half, 32), (short)0, a, false, false);
                acc[g] = a;
            }
            v8f c = c1[j];
#pragma unroll
            for (int r = 0; r < 8; ++r) {
                const float iv = sigf(acc[0][r]);
                const float fv = sigf(acc[1][r]);
                const float gv = vtanh(acc[2][r]);
                const float ov = sigf(acc[3][r]);
                const float cv = fv * c[r] + iv * gv;
                c[r] = cv;
                const float hv = ov * vtanh(cv);
                const int m = r + 8 * half;
                h1buf[m * HH + j * 16 + li] = (_Float16)hv;
            }
            c1[j] = c;
        }
        asm volatile("s_wait_dscnt 0" ::: "memory");
        a1lo = lds_a16(h1buf, li, half, 0);
        a1hi = lds_a16(h1buf, li, half, 32);

        // ================= Layer 2 =================
        const bool is_out = ((t % 5) == 4);
        const int  sidx   = t / 5;
#pragma unroll
        for (int j = 0; j < 4; ++j) {
            v8f acc[4];
#pragma unroll
            for (int g = 0; g < 4; ++g) {
                const int n0 = g * HH + j * 16;
                const float bv = sB1[n0 + li];
                v8f a = {bv, bv, bv, bv, bv, bv, bv, bv};
                a = __builtin_amdgcn_wmma_f32_16x16x32_f16(false, a1lo, false,
                        lds_b16(sWih1, n0, li, half, 0), (short)0, a, false, false);
                a = __builtin_amdgcn_wmma_f32_16x16x32_f16(false, a1hi, false,
                        lds_b16(sWih1, n0, li, half, 32), (short)0, a, false, false);
                a = __builtin_amdgcn_wmma_f32_16x16x32_f16(false, a2lo, false,
                        lds_b16(sWhh1, n0, li, half, 0), (short)0, a, false, false);
                a = __builtin_amdgcn_wmma_f32_16x16x32_f16(false, a2hi, false,
                        lds_b16(sWhh1, n0, li, half, 32), (short)0, a, false, false);
                acc[g] = a;
            }
            v8f c = c2[j];
#pragma unroll
            for (int r = 0; r < 8; ++r) {
                const float iv = sigf(acc[0][r]);
                const float fv = sigf(acc[1][r]);
                const float gv = vtanh(acc[2][r]);
                const float ov = sigf(acc[3][r]);
                const float cv = fv * c[r] + iv * gv;
                c[r] = cv;
                const float hv = ov * vtanh(cv);
                const int m = r + 8 * half;
                h2buf[m * HH + j * 16 + li] = (_Float16)hv;
                if (is_out) {
                    out[((size_t)(brow + m) * 5 + sidx) * HH + j * 16 + li] = hv;
                }
            }
            c2[j] = c;
        }
        asm volatile("s_wait_dscnt 0" ::: "memory");
        a2lo = lds_a16(h2buf, li, half, 0);
        a2hi = lds_a16(h2buf, li, half, 32);
    }
}

extern "C" void kernel_launch(void* const* d_in, const int* in_sizes, int n_in,
                              void* d_out, int out_size, void* d_ws, size_t ws_size,
                              hipStream_t stream) {
    const float* x    = (const float*)d_in[0];
    const float* Wih0 = (const float*)d_in[1];
    const float* Whh0 = (const float*)d_in[2];
    const float* b0g  = (const float*)d_in[3];
    const float* Wih1 = (const float*)d_in[4];
    const float* Whh1 = (const float*)d_in[5];
    const float* b1g  = (const float*)d_in[6];
    float* out = (float*)d_out;

    const int B = in_sizes[0] / (TT * II);   // 65536
    const int grid = B / ROWS_PER_BLOCK;     // 512

    lstm2_wmma_kernel<<<grid, 256, 0, stream>>>(x, Wih0, Whh0, b0g, Wih1, Whh1, b1g, out);
}
